// Model_65953517797994
// MI455X (gfx1250) — matrix operand order repaired and checked by reading, not compiled, not run on hardware
//
#include <hip/hip_runtime.h>

#define DI __device__ __forceinline__

typedef __attribute__((ext_vector_type(16))) __bf16   v16bf;
typedef __attribute__((ext_vector_type(8)))  float    v8f;
typedef __attribute__((ext_vector_type(4)))  float    f32x4;
typedef __attribute__((ext_vector_type(4)))  unsigned u32x4;
typedef __attribute__((ext_vector_type(4)))  int      v4i;

#if defined(__AMDGCN__) && __has_builtin(__builtin_amdgcn_global_load_async_to_lds_b128) && __has_builtin(__builtin_amdgcn_s_wait_asynccnt)
#define HAS_ASYNC 1
typedef __attribute__((address_space(1))) v4i as1_v4i;   // global int4*
typedef __attribute__((address_space(3))) v4i as3_v4i;   // LDS int4*
#else
#define HAS_ASYNC 0
#endif

constexpr int BATCH  = 16384;
constexpr int L      = 8192;
constexpr int L3     = 8189;   // length after 3 valid K=2 convs
constexpr int NOUT   = 128;
constexpr int KPAD   = 8192;   // L3 padded to multiple of 64 (zeros)
constexpr int BM     = 64;     // batch rows per block
constexpr int BN     = 128;    // all OUT columns per block
constexpr int KC     = 64;     // K-chunk per LDS stage
constexpr int LSTR   = 72;     // LDS row stride in bf16 (KC + 8 pad)
constexpr int NCHUNK = KPAD / KC;   // 128

// f32 -> bf16 via native fptrunc (RNE; backend can fuse to v_cvt_pk_bf16_f32)
DI unsigned short f2bf(float f) { return __builtin_bit_cast(unsigned short, (__bf16)f); }
DI unsigned packbf(float lo, float hi) {
  return (unsigned)f2bf(lo) | ((unsigned)f2bf(hi) << 16);
}
DI float relu(float v) { return fmaxf(v, 0.0f); }

union Frag { u32x4 q[2]; v16bf v; };
struct BRegs { u32x4 q[4]; };

DI void wait_async() {
#if HAS_ASYNC
  __builtin_amdgcn_s_wait_asynccnt(0);
#endif
}

// ---------------------------------------------------------------------------
// One-time W f32 -> bf16 pre-conversion into workspace, zero-padded to KPAD.
// ---------------------------------------------------------------------------
__global__ __launch_bounds__(256) void convert_w_bf16(
    const float* __restrict__ Wm, unsigned short* __restrict__ Wb)
{
  const int tid = blockIdx.x * 256 + threadIdx.x;  // 131072 threads total
  const int n = tid >> 10;                          // output row (0..127)
  const int q = tid & 1023;                         // u32x4 index in row
  const int k = q * 8;
  const float* src = Wm + (size_t)n * L3;
  float v[8];
  #pragma unroll
  for (int i = 0; i < 8; ++i) v[i] = (k + i < L3) ? src[k + i] : 0.f;
  u32x4 pk = {packbf(v[0], v[1]), packbf(v[2], v[3]),
              packbf(v[4], v[5]), packbf(v[6], v[7])};
  *(u32x4*)(Wb + (size_t)n * KPAD + k) = pk;
}

// ---------------------------------------------------------------------------
// Pipeline stage 1a: issue x loads for a chunk (NT, streamed once)
// ---------------------------------------------------------------------------
template <bool GUARD>
DI void load_x(int k0, int aseg, const float* __restrict__ xr, float (&xv)[20])
{
  const int j0 = k0 + aseg * 16;
  #pragma unroll
  for (int i = 0; i < 5; ++i) {
    const int idx = j0 + i * 4;
    f32x4 v = {0.f, 0.f, 0.f, 0.f};
    if (!GUARD || idx < L)
      v = __builtin_nontemporal_load((const f32x4*)(xr + idx));
    xv[i*4+0] = v.x; xv[i*4+1] = v.y; xv[i*4+2] = v.z; xv[i*4+3] = v.w;
  }
  if (!GUARD) __builtin_prefetch(xr + j0 + KC, 0, 0);
}

// ---------------------------------------------------------------------------
// Pipeline stage 1b: B tile (bf16 W) -> LDS, async DMA when available
// ---------------------------------------------------------------------------
template <bool WB16>
DI void load_b(int k0, int bn, int bseg,
               const unsigned short* __restrict__ wbr,
               unsigned short (*Bs)[LSTR], BRegs& brg)
{
  if (!WB16) return;                       // f32 fallback handled in store phase
  const int j0b = k0 + bseg * 32;
#if HAS_ASYNC
  void* gpv = (void*)(wbr + j0b);          // 64B per thread, 16B per op
  as1_v4i* gp = (as1_v4i*)gpv;             // global side: typed int4* (as1)
  as3_v4i* lp = (as3_v4i*)&Bs[bn][bseg * 32];  // LDS side: typed int4* (as3)
  __builtin_amdgcn_global_load_async_to_lds_b128(gp, lp,  0, 0);
  __builtin_amdgcn_global_load_async_to_lds_b128(gp, lp, 16, 0);
  __builtin_amdgcn_global_load_async_to_lds_b128(gp, lp, 32, 0);
  __builtin_amdgcn_global_load_async_to_lds_b128(gp, lp, 48, 0);
  (void)brg;
#else
  const u32x4* src = (const u32x4*)(wbr + j0b);
  brg.q[0] = src[0]; brg.q[1] = src[1]; brg.q[2] = src[2]; brg.q[3] = src[3];
#endif
}

// ---------------------------------------------------------------------------
// Pipeline stage 2: conv chain + pack -> A tile; flush staged B regs
// ---------------------------------------------------------------------------
template <bool GUARD, bool WB16>
DI void store_tiles(int k0, int ar, int aseg, float (&xv)[20],
                    int bn, int bseg, const float* __restrict__ wr, BRegs& brg,
                    unsigned short (*As)[LSTR], unsigned short (*Bs)[LSTR],
                    float w10, float w11, float b1v,
                    float w20, float w21, float b2v,
                    float w30, float w31, float b3v)
{
  const int j0 = k0 + aseg * 16;
  #pragma unroll
  for (int i = 0; i < 19; ++i)
    xv[i] = relu(fmaf(w11, xv[i+1], fmaf(w10, xv[i], b1v)));
  #pragma unroll
  for (int i = 0; i < 18; ++i)
    xv[i] = relu(fmaf(w21, xv[i+1], fmaf(w20, xv[i], b2v)));
  #pragma unroll
  for (int i = 0; i < 16; ++i)
    xv[i] = relu(fmaf(w31, xv[i+1], fmaf(w30, xv[i], b3v)));
  if (GUARD && (j0 + 15 >= L3)) {          // zero K padding, last chunk only
    #pragma unroll
    for (int i = 0; i < 16; ++i) if (j0 + i >= L3) xv[i] = 0.f;
  }
  {
    u32x4* dst = (u32x4*)&As[ar][aseg * 16];
    dst[0] = u32x4{packbf(xv[ 0], xv[ 1]), packbf(xv[ 2], xv[ 3]),
                   packbf(xv[ 4], xv[ 5]), packbf(xv[ 6], xv[ 7])};
    dst[1] = u32x4{packbf(xv[ 8], xv[ 9]), packbf(xv[10], xv[11]),
                   packbf(xv[12], xv[13]), packbf(xv[14], xv[15])};
  }

  const int j0b = k0 + bseg * 32;
  if (WB16) {
#if !HAS_ASYNC
    u32x4* dst = (u32x4*)&Bs[bn][bseg * 32];
    dst[0] = brg.q[0]; dst[1] = brg.q[1]; dst[2] = brg.q[2]; dst[3] = brg.q[3];
#else
    (void)brg; (void)wr;
#endif
  } else {
    unsigned pk[16];
    if (!GUARD || (j0b + 31 < L3)) {
      #pragma unroll
      for (int i = 0; i < 16; ++i)
        pk[i] = packbf(wr[j0b + 2*i], wr[j0b + 2*i + 1]);
    } else {
      #pragma unroll
      for (int i = 0; i < 16; ++i) {
        const int j = j0b + 2*i;
        const float lo = (j     < L3) ? wr[j]     : 0.f;
        const float hh = (j + 1 < L3) ? wr[j + 1] : 0.f;
        pk[i] = packbf(lo, hh);
      }
    }
    u32x4* dst = (u32x4*)&Bs[bn][bseg * 32];
    dst[0] = u32x4{pk[ 0], pk[ 1], pk[ 2], pk[ 3]};
    dst[1] = u32x4{pk[ 4], pk[ 5], pk[ 6], pk[ 7]};
    dst[2] = u32x4{pk[ 8], pk[ 9], pk[10], pk[11]};
    dst[3] = u32x4{pk[12], pk[13], pk[14], pk[15]};
  }
}

// ---------------------------------------------------------------------------
// Consume one staged chunk: 2 K-steps of 32, 4 WMMAs each (32x32 per wave)
// ---------------------------------------------------------------------------
DI void consume_tiles(unsigned short (*As)[LSTR], unsigned short (*Bs)[LSTR],
                      int waveM, int waveN, int lr, int hi, v8f (&acc)[2][2])
{
  #pragma unroll
  for (int kk = 0; kk < KC; kk += 32) {
    Frag a[2], b[2];
    #pragma unroll
    for (int tm = 0; tm < 2; ++tm) {
      // A lane layout: half 0 -> K {0..7,16..23}, half 1 -> K {8..15,24..31}
      const unsigned short* p = &As[waveM*32 + tm*16 + lr][kk + hi*8];
      a[tm].q[0] = *(const u32x4*)(p);
      a[tm].q[1] = *(const u32x4*)(p + 16);
    }
    #pragma unroll
    for (int tn = 0; tn < 2; ++tn) {
      // B lane layout: half 0 -> K {0..7,16..23}, half 1 -> K {8..15,24..31}
      const unsigned short* p = &Bs[waveN*32 + tn*16 + lr][kk + hi*8];
      b[tn].q[0] = *(const u32x4*)(p);
      b[tn].q[1] = *(const u32x4*)(p + 16);
    }
    #pragma unroll
    for (int tm = 0; tm < 2; ++tm)
      #pragma unroll
      for (int tn = 0; tn < 2; ++tn)
        acc[tm][tn] = __builtin_amdgcn_wmma_f32_16x16x32_bf16(
            false, a[tm].v, false, b[tn].v,
            (short)0, acc[tm][tn], false, false);
  }
}

// ---------------------------------------------------------------------------
// Fused conv3 + bf16 WMMA GEMM, double-buffered software pipeline
// ---------------------------------------------------------------------------
template <bool WB16>
__global__ __launch_bounds__(256) void fused_conv3_wmma_gemm(
    const float* __restrict__ x,
    const float* __restrict__ w1, const float* __restrict__ b1,
    const float* __restrict__ w2, const float* __restrict__ b2,
    const float* __restrict__ w3, const float* __restrict__ b3,
    const float* __restrict__ Wm, const unsigned short* __restrict__ Wb,
    const float* __restrict__ bias, float* __restrict__ out)
{
  __shared__ unsigned short As[2][BM][LSTR];
  __shared__ unsigned short Bs[2][BN][LSTR];

  const int t     = threadIdx.x;
  const int wave  = t >> 5, lane = t & 31;
  const int waveM = wave & 1;          // 0..1 -> 32-row strip
  const int waveN = wave >> 1;         // 0..3 -> 32-col strip
  const int lr    = lane & 15;
  const int hi    = lane >> 4;

  const float w10 = w1[0], w11 = w1[1], b1v = b1[0];
  const float w20 = w2[0], w21 = w2[1], b2v = b2[0];
  const float w30 = w3[0], w31 = w3[1], b3v = b3[0];

  const int ar = t & 63, aseg = t >> 6;                  // A: 64 rows x 4 segs
  const float* xr = x + (size_t)(blockIdx.x * BM + ar) * L;
  const int bn = t >> 1, bseg = t & 1;                   // B: 128 rows x 2 segs
  const float*          wr  = Wm + (size_t)bn * L3;
  const unsigned short* wbr = WB16 ? (Wb + (size_t)bn * KPAD) : nullptr;

  v8f acc[2][2] = {};
  float xv[20];
  BRegs brg;

  // prologue: stage chunk 0 into buffer 0
  load_x<false>(0, aseg, xr, xv);
  load_b<WB16>(0, bn, bseg, wbr, Bs[0], brg);
  store_tiles<false, WB16>(0, ar, aseg, xv, bn, bseg, wr, brg, As[0], Bs[0],
                           w10, w11, b1v, w20, w21, b2v, w30, w31, b3v);
  wait_async();
  __syncthreads();

  #pragma unroll 1
  for (int c = 0; c < NCHUNK - 2; ++c) {                 // c = 0..125
    const int k0n = (c + 1) * KC;
    load_x<false>(k0n, aseg, xr, xv);                    // chunk c+1 in flight
    load_b<WB16>(k0n, bn, bseg, wbr, Bs[(c + 1) & 1], brg);
    consume_tiles(As[c & 1], Bs[c & 1], waveM, waveN, lr, hi, acc);
    store_tiles<false, WB16>(k0n, ar, aseg, xv, bn, bseg, wr, brg,
                             As[(c + 1) & 1], Bs[(c + 1) & 1],
                             w10, w11, b1v, w20, w21, b2v, w30, w31, b3v);
    wait_async();
    __syncthreads();
  }
  {  // c = 126: consume buf0, stage guarded final chunk 127 into buf1
    const int k0n = (NCHUNK - 1) * KC;
    load_x<true>(k0n, aseg, xr, xv);
    load_b<WB16>(k0n, bn, bseg, wbr, Bs[1], brg);
    consume_tiles(As[0], Bs[0], waveM, waveN, lr, hi, acc);
    store_tiles<true, WB16>(k0n, ar, aseg, xv, bn, bseg, wr, brg, As[1], Bs[1],
                            w10, w11, b1v, w20, w21, b2v, w30, w31, b3v);
    wait_async();
    __syncthreads();
  }
  consume_tiles(As[1], Bs[1], waveM, waveN, lr, hi, acc);  // final chunk

  // ---- epilogue: bias add + streaming f32 stores ----
  const int rowBase = blockIdx.x * BM + waveM * 32 + hi * 8;
  #pragma unroll
  for (int tn = 0; tn < 2; ++tn) {
    const int col = waveN * 32 + tn * 16 + lr;
    const float bv = bias[col];
    #pragma unroll
    for (int tm = 0; tm < 2; ++tm) {
      const int r0 = rowBase + tm * 16;
      #pragma unroll
      for (int i = 0; i < 8; ++i)
        __builtin_nontemporal_store(acc[tm][tn][i] + bv,
                                    out + (size_t)(r0 + i) * NOUT + col);
    }
  }
}

extern "C" void kernel_launch(void* const* d_in, const int* in_sizes, int n_in,
                              void* d_out, int out_size, void* d_ws, size_t ws_size,
                              hipStream_t stream) {
  (void)in_sizes; (void)n_in; (void)out_size;
  const float* x    = (const float*)d_in[0];
  const float* w1   = (const float*)d_in[1];
  const float* b1   = (const float*)d_in[2];
  const float* w2   = (const float*)d_in[3];
  const float* b2   = (const float*)d_in[4];
  const float* w3   = (const float*)d_in[5];
  const float* b3   = (const float*)d_in[6];
  const float* Wm   = (const float*)d_in[7];
  const float* bias = (const float*)d_in[8];
  float* out        = (float*)d_out;

  const size_t wb_bytes = (size_t)NOUT * KPAD * sizeof(unsigned short); // 2 MiB
  dim3 grid(BATCH / BM);   // 256 blocks
  dim3 block(256);         // 8 waves (wave32)

  if (ws_size >= wb_bytes) {
    unsigned short* Wb = (unsigned short*)d_ws;
    convert_w_bf16<<<dim3((NOUT * KPAD / 8) / 256), 256, 0, stream>>>(Wm, Wb);
    fused_conv3_wmma_gemm<true><<<grid, block, 0, stream>>>(
        x, w1, b1, w2, b2, w3, b3, Wm, Wb, bias, out);
  } else {
    fused_conv3_wmma_gemm<false><<<grid, block, 0, stream>>>(
        x, w1, b1, w2, b2, w3, b3, Wm, nullptr, bias, out);
  }
}
